// SRUCell_81441169866771
// MI455X (gfx1250) — matrix operand order repaired and checked by reading, not compiled, not run on hardware
//
#include <hip/hip_runtime.h>

typedef __attribute__((ext_vector_type(16))) __bf16 v16bf;
typedef __attribute__((ext_vector_type(8)))  float  v8f;
typedef __attribute__((ext_vector_type(4)))  float  fvec4;
typedef __attribute__((ext_vector_type(2)))  unsigned uvec2;

constexpr int L  = 1024;
constexpr int B  = 32;
constexpr int IN = 1024;
constexpr int D  = 1024;
constexpr int M  = L * B;    // 32768 rows of the GEMM
constexpr int K  = IN;       // 1024
constexpr int N  = 4 * D;    // 4096 cols

constexpr int BM = 128;      // workgroup tile M
constexpr int BN = 128;      // workgroup tile N
constexpr int BK = 32;       // K step == WMMA K
constexpr int NKB = K / BK;  // 32 K-blocks
constexpr int LDSS = 48;     // padded LDS row stride (bf16): 96B rows, 16B-aligned

// pack two fp32 -> packed bf16x2 (round-to-nearest-even), a in [15:0], b in [31:16]
static __device__ __forceinline__ unsigned pkbf2(float a, float b) {
  unsigned ua = __builtin_bit_cast(unsigned, a);
  unsigned ub = __builtin_bit_cast(unsigned, b);
  ua += 0x7FFFu + ((ua >> 16) & 1u);
  ub += 0x7FFFu + ((ub >> 16) & 1u);
  return (ua >> 16) | (ub & 0xFFFF0000u);
}

// -------------------- Phase 1: U = X(32768x1024) * W(1024x4096), bf16 WMMA --------------------
__global__ __launch_bounds__(256) void sru_gemm_wmma(const float* __restrict__ X,
                                                     const float* __restrict__ Wm,
                                                     float* __restrict__ U) {
  __shared__ __bf16 As[2][BM][LDSS];   // A panels, row-major [m][k]
  __shared__ __bf16 Bs[2][BN][LDSS];   // B panels, transposed [n][k]

  const int t     = threadIdx.x;
  const int lane  = t & 31;
  const int wave  = t >> 5;      // 0..7
  const int waveM = wave & 3;    // 4 waves along M -> 32 rows each
  const int waveN = wave >> 2;   // 2 waves along N -> 64 cols each

  const int blockN = blockIdx.x * BN;
  const int blockM = blockIdx.y * BM;

  // staging assignments (256 threads)
  const int ar = t >> 3;          // 0..31  A row within quarter-panel
  const int ak = (t & 7) * 4;     // 0..28  A k (float4 granularity)
  const int bn = (t & 31) * 4;    // 0..124 B col (float4 granularity)
  const int bkp = (t >> 5) * 2;   // 0..14  B k-pair base (even k)

  // stage one 128x32 A panel + 32x128 B panel into LDS buffer `buf` (fp32 -> bf16)
  auto stage = [&](int buf, int k0) {
    #pragma unroll
    for (int i = 0; i < 4; ++i) {
      const int r = ar + i * 32;
      const fvec4 v = *(const fvec4*)(X + (size_t)(blockM + r) * K + k0 + ak);
      uvec2 p;
      p[0] = pkbf2(v[0], v[1]);
      p[1] = pkbf2(v[2], v[3]);
      *(uvec2*)(&As[buf][r][ak]) = p;          // one ds_store_b64
    }
    #pragma unroll
    for (int i = 0; i < 2; ++i) {
      const int kk = bkp + i * 16;             // even k
      const fvec4 v0 = *(const fvec4*)(Wm + (size_t)(k0 + kk)     * N + blockN + bn);
      const fvec4 v1 = *(const fvec4*)(Wm + (size_t)(k0 + kk + 1) * N + blockN + bn);
      #pragma unroll
      for (int c = 0; c < 4; ++c) {
        *(unsigned*)(&Bs[buf][bn + c][kk]) = pkbf2(v0[c], v1[c]);  // ds_store_b32
      }
    }
  };

  v8f acc[2][4] = {};

  // fragment addressing (contiguous k = 16*(lane/16) + s for both A and B)
  const int rsel = lane & 15;
  const int kh   = (lane >> 4) << 4;   // A: lanes<16 -> K0..15; lanes>=16 -> K16..31
  const int kb   = (lane >> 4) << 4;   // B: lanes<16 -> K0..15; lanes>=16 -> K16..31

  stage(0, 0);
  __syncthreads();

  for (int kblk = 0; kblk < NKB; ++kblk) {
    const int cur = kblk & 1;
    if (kblk + 1 < NKB) stage(cur ^ 1, (kblk + 1) * BK);  // overlap next panel with this step's math

    v16bf afrag[2];
    #pragma unroll
    for (int mt = 0; mt < 2; ++mt) {
      const __bf16* src = &As[cur][waveM * 32 + mt * 16 + rsel][0];
      #pragma unroll
      for (int j = 0; j < 8; ++j) {
        afrag[mt][j]     = src[kh + j];
        afrag[mt][8 + j] = src[kh + 8 + j];
      }
    }
    v16bf bfrag[4];
    #pragma unroll
    for (int nt = 0; nt < 4; ++nt) {
      const __bf16* src = &Bs[cur][waveN * 64 + nt * 16 + rsel][0];
      #pragma unroll
      for (int j = 0; j < 16; ++j) bfrag[nt][j] = src[kb + j];
    }

    #pragma unroll
    for (int mt = 0; mt < 2; ++mt) {
      #pragma unroll
      for (int nt = 0; nt < 4; ++nt) {
        acc[mt][nt] = __builtin_amdgcn_wmma_f32_16x16x32_bf16(
            false, afrag[mt], false, bfrag[nt], (short)0, acc[mt][nt], false, false);
      }
    }
    __syncthreads();  // frag reads consumed above (dscnt waited) before signaling
  }

  // epilogue: element e of v8f is row e (+8 for upper lanes), col = lane&15.
  // U is streamed (512 MB, read once by the scan) -> non-temporal stores.
  const int rowHalf = (lane >> 4) << 3;
  #pragma unroll
  for (int mt = 0; mt < 2; ++mt) {
    #pragma unroll
    for (int nt = 0; nt < 4; ++nt) {
      const int gn    = blockN + waveN * 64 + nt * 16 + (lane & 15);
      const int rbase = blockM + waveM * 32 + mt * 16 + rowHalf;
      #pragma unroll
      for (int e = 0; e < 8; ++e) {
        __builtin_nontemporal_store(acc[mt][nt][e], U + (size_t)(rbase + e) * N + gn);
      }
    }
  }
}

// -------------------- Phase 2: sequential SRU scan over L --------------------
__global__ __launch_bounds__(256) void sru_scan(const fvec4* __restrict__ U4,
                                                const float* __restrict__ c0,
                                                const float* __restrict__ V,
                                                const float* __restrict__ bias,
                                                float* __restrict__ h,
                                                float* __restrict__ c_last) {
  const int idx = blockIdx.x * blockDim.x + threadIdx.x;  // 0 .. B*D-1
  const int b = idx >> 10;          // / D
  const int d = idx & (D - 1);      // % D

  float c = c0[idx];
  const float vf = V[d];
  const float vr = V[D + d];
  const float bf_ = bias[d];
  const float br_ = bias[D + d];

  // U row = 4096 floats = 1024 fvec4; gates for feature d are fvec4 #d
  fvec4 u = __builtin_nontemporal_load(U4 + (size_t)b * (N / 4) + d);  // l = 0
  for (int l = 0; l < L; ++l) {
    fvec4 un = u;
    if (l + 1 < L)
      un = __builtin_nontemporal_load(U4 + (size_t)((l + 1) * B + b) * (N / 4) + d);

    const float f = 1.0f / (1.0f + __expf(-(u[1] + c * vf + bf_)));
    const float r = 1.0f / (1.0f + __expf(-(u[2] + c * vr + br_)));
    c = u[0] + (c - u[0]) * f;
    const float e2 = __expf(-2.0f * c);
    const float th = (1.0f - e2) / (1.0f + e2);  // tanh(c)
    __builtin_nontemporal_store((th - u[3]) * r + u[3],
                                h + (size_t)(l * B + b) * D + d);
    u = un;
  }
  c_last[idx] = c;
}

extern "C" void kernel_launch(void* const* d_in, const int* in_sizes, int n_in,
                              void* d_out, int out_size, void* d_ws, size_t ws_size,
                              hipStream_t stream) {
  (void)in_sizes; (void)n_in; (void)out_size; (void)ws_size;
  const float* x    = (const float*)d_in[0];  // (L,B,IN)  == (M,K)
  const float* c0   = (const float*)d_in[1];  // (B,D)
  const float* Wm   = (const float*)d_in[2];  // (IN,4D)   == (K,N)
  const float* V    = (const float*)d_in[3];  // (2D,)
  const float* bias = (const float*)d_in[4];  // (2D,)

  float* U      = (float*)d_ws;               // M*N fp32 scratch (512 MB)
  float* h      = (float*)d_out;              // (L,B,D)
  float* c_last = h + (size_t)L * B * D;      // (B,D) appended after h

  sru_gemm_wmma<<<dim3(N / BN, M / BM), 256, 0, stream>>>(x, Wm, U);
  sru_scan<<<(B * D) / 256, 256, 0, stream>>>((const fvec4*)U, c0, V, bias, h, c_last);
}
